// GGNN_LCG_84370337563244
// MI455X (gfx1250) — matrix operand order repaired and checked by reading, not compiled, not run on hardware
//
#include <hip/hip_runtime.h>

#define DMODEL 128
#define N_ITER 8

typedef __bf16 v16bf __attribute__((ext_vector_type(16)));
typedef float  v8f   __attribute__((ext_vector_type(8)));

struct __align__(16) u128 { unsigned int x0, x1, x2, x3; };
struct F2 { u128 lo, hi; };

static __device__ __forceinline__ unsigned short f2bf(float f) {
  union { float f; unsigned int u; } v; v.f = f;
  unsigned int r = v.u + 0x7fffu + ((v.u >> 16) & 1u);
  return (unsigned short)(r >> 16);
}

static __device__ __forceinline__ v16bf load_frag(const unsigned short* p0,
                                                  const unsigned short* p1) {
  F2 t;
  t.lo = *(const u128*)p0;
  t.hi = *(const u128*)p1;
  return __builtin_bit_cast(v16bf, t);
}

static __device__ __forceinline__ v8f wmma_bf16(v16bf a, v16bf b, v8f c) {
  return __builtin_amdgcn_wmma_f32_16x16x32_bf16(false, a, false, b,
                                                 (short)0, c, false, false);
}

static __device__ __forceinline__ float sigm(float x) {
  return 1.0f / (1.0f + __expf(-x));
}
static __device__ __forceinline__ float tanh_fast(float x) {
  float e = __expf(-2.0f * x);
  return (1.0f - e) / (1.0f + e);
}

// ---------------------------------------------------------------------------
// Pack fp32 weight W[N,K] (row-major, out = x @ W.T) into WMMA B-fragment
// layout: fragment (nt,kt) stored as 32 lanes x 16 bf16 halves.
// Lane l holds column n = nt*16 + (l&15), K values kt*32 + (l>>4)*8 + j (j<8), +16 + (j-8) (j>=8).
// Fragment index = nt*KT + kt (inner K loop streams contiguously).
// ---------------------------------------------------------------------------
__global__ void pack_weight(const float* __restrict__ W,
                            unsigned short* __restrict__ out,
                            int NT, int KT, int K) {
  int gid = blockIdx.x * blockDim.x + threadIdx.x;
  int total = NT * KT * 32;
  if (gid >= total) return;
  int lane = gid & 31;
  int frag = gid >> 5;
  int kt = frag % KT;
  int nt = frag / KT;
  int n = nt * 16 + (lane & 15);
  int kb = kt * 32 + ((lane >> 4) << 3);
  const float* wr = W + (size_t)n * K + kb;
  unsigned short* o = out + (size_t)frag * 512 + lane * 16;
#pragma unroll
  for (int j = 0; j < 16; ++j) o[j] = f2bf(wr[j < 8 ? j : j + 8]);
}

// ---------------------------------------------------------------------------
// fp32 rows (128 cols) -> bf16 rows; output row index can be XORed (literal
// pos/neg swap) and output can have a different leading dimension (for the
// concatenated [c2l_aggr | l2l_msg] GRU input).
// ---------------------------------------------------------------------------
__global__ void cvt_rows_bf16(const float* __restrict__ in, int ldi,
                              unsigned short* __restrict__ out, int ldo,
                              int rows, int xr) {
  int gid = blockIdx.x * blockDim.x + threadIdx.x;
  if (gid >= rows * 32) return;
  int r = gid >> 5;
  int c0 = (gid & 31) * 4;
  float4 v = *(const float4*)(in + (size_t)r * ldi + c0);
  unsigned short* o = out + (size_t)(r ^ xr) * ldo + c0;
  o[0] = f2bf(v.x); o[1] = f2bf(v.y); o[2] = f2bf(v.z); o[3] = f2bf(v.w);
}

// ---------------------------------------------------------------------------
// GEMM: out[M,128] = act(A[M,128](bf16) @ Bpacked + bias). One wave per
// 16-row strip; A fragments held in registers across all 8 N-tiles.
// B fragments are a flat sequential stream of 32 fragments -> software
// pipelined: load fragment f+1 before the WMMA consuming fragment f, so the
// next load overlaps the current WMMA instead of a full s_wait_loadcnt 0.
// ---------------------------------------------------------------------------
template<bool RELU, bool OUTBF>
__global__ __launch_bounds__(256)
void gemm128_bf16(const unsigned short* __restrict__ A, int lda,
                  const unsigned short* __restrict__ Bp,
                  const float* __restrict__ bias,
                  float* __restrict__ outF, unsigned short* __restrict__ outB,
                  int ldo, int M) {
  const int wave = threadIdx.x >> 5;
  const int lane = threadIdx.x & 31;
  const int m0 = (blockIdx.x * 8 + wave) * 16;
  if (m0 >= M) return;
  const int l15 = lane & 15, hi16 = lane >> 4;
  const unsigned short* arow = A + (size_t)(m0 + l15) * lda;
  v16bf a[4];
#pragma unroll
  for (int kt = 0; kt < 4; ++kt) {
    const int k0 = kt * 32 + 8 * hi16;   // 16-bit A 16x32 lane layout
    a[kt] = load_frag(arow + k0, arow + k0 + 16);
  }
  const unsigned short* bp = Bp + lane * 16;
  v16bf bcur = load_frag(bp, bp + 8);
  const int n = l15;
#pragma unroll
  for (int nt = 0; nt < 8; ++nt) {
    v8f acc = {};
#pragma unroll
    for (int kt = 0; kt < 4; ++kt) {
      v16bf bnext;
      if (!(nt == 7 && kt == 3)) {       // prefetch next fragment in stream
        bp += 512;
        bnext = load_frag(bp, bp + 8);
      }
      acc = wmma_bf16(a[kt], bcur, acc);
      if (!(nt == 7 && kt == 3)) bcur = bnext;
    }
    const int nc = nt * 16 + n;
    const float bv = bias[nc];
#pragma unroll
    for (int v = 0; v < 8; ++v) {        // C layout: row = m0 + v + 8*hi
      float x = acc[v] + bv;
      if (RELU) x = fmaxf(x, 0.0f);
      const size_t o = (size_t)(m0 + v + 8 * hi16) * ldo + nc;
      if (OUTBF) outB[o] = f2bf(x); else outF[o] = x;
    }
  }
}

// ---------------------------------------------------------------------------
// Fused GRUCell: computes all 6 gate pre-activations via WMMA (gi = x@Wih.T,
// gh = h@Whh.T, N=384 split as r/z/n tiles dt, dt+8, dt+16), then applies
// sigmoid/tanh gating in-register. gi/gh are never materialized.
// ---------------------------------------------------------------------------
template<int KXT>   // K of x input / 32 : 4 (clause GRU) or 8 (literal GRU)
__global__ __launch_bounds__(256)
void gru_fused(const unsigned short* __restrict__ X, int ldx,
               const unsigned short* __restrict__ H,       // ld = 128
               const unsigned short* __restrict__ WihP,    // NT=24, KT=KXT
               const unsigned short* __restrict__ WhhP,    // NT=24, KT=4
               const float* __restrict__ bih, const float* __restrict__ bhh,
               const float* __restrict__ hOld, float* __restrict__ outNew,
               int M) {
  const int wave = threadIdx.x >> 5;
  const int lane = threadIdx.x & 31;
  const int m0 = (blockIdx.x * 8 + wave) * 16;
  if (m0 >= M) return;
  const int l15 = lane & 15, hi16 = lane >> 4;
  const unsigned short* xrow = X + (size_t)(m0 + l15) * ldx;
  const unsigned short* hrow = H + (size_t)(m0 + l15) * DMODEL;
  v16bf ax[KXT], ah[4];
#pragma unroll
  for (int kt = 0; kt < KXT; ++kt) {
    const int k0 = kt * 32 + 8 * hi16;
    ax[kt] = load_frag(xrow + k0, xrow + k0 + 16);
  }
#pragma unroll
  for (int kt = 0; kt < 4; ++kt) {
    const int k0 = kt * 32 + 8 * hi16;
    ah[kt] = load_frag(hrow + k0, hrow + k0 + 16);
  }
#pragma unroll
  for (int dt = 0; dt < 8; ++dt) {
    v8f ir = {}, iz = {}, inn = {}, hr = {}, hz = {}, hn = {};
#pragma unroll
    for (int kt = 0; kt < KXT; ++kt) {
      const unsigned short* b0 = WihP + (size_t)((dt)      * KXT + kt) * 512 + lane * 16;
      const unsigned short* b1 = WihP + (size_t)((dt + 8)  * KXT + kt) * 512 + lane * 16;
      const unsigned short* b2 = WihP + (size_t)((dt + 16) * KXT + kt) * 512 + lane * 16;
      v16bf f0 = load_frag(b0, b0 + 8);   // issue all three loads first so the
      v16bf f1 = load_frag(b1, b1 + 8);   // compiler clauses them and overlaps
      v16bf f2 = load_frag(b2, b2 + 8);   // them with the WMMA burst
      ir  = wmma_bf16(ax[kt], f0, ir);
      iz  = wmma_bf16(ax[kt], f1, iz);
      inn = wmma_bf16(ax[kt], f2, inn);
    }
#pragma unroll
    for (int kt = 0; kt < 4; ++kt) {
      const unsigned short* b0 = WhhP + (size_t)((dt)      * 4 + kt) * 512 + lane * 16;
      const unsigned short* b1 = WhhP + (size_t)((dt + 8)  * 4 + kt) * 512 + lane * 16;
      const unsigned short* b2 = WhhP + (size_t)((dt + 16) * 4 + kt) * 512 + lane * 16;
      v16bf f0 = load_frag(b0, b0 + 8);
      v16bf f1 = load_frag(b1, b1 + 8);
      v16bf f2 = load_frag(b2, b2 + 8);
      hr = wmma_bf16(ah[kt], f0, hr);
      hz = wmma_bf16(ah[kt], f1, hz);
      hn = wmma_bf16(ah[kt], f2, hn);
    }
    const int n = dt * 16 + l15;
    const float bir = bih[n], biz = bih[n + 128], bin = bih[n + 256];
    const float bhr = bhh[n], bhz = bhh[n + 128], bhn = bhh[n + 256];
#pragma unroll
    for (int v = 0; v < 8; ++v) {
      const size_t idx = (size_t)(m0 + v + 8 * hi16) * DMODEL + n;
      const float h0 = hOld[idx];
      const float r = sigm(ir[v] + bir + hr[v] + bhr);
      const float z = sigm(iz[v] + biz + hz[v] + bhz);
      const float nn = tanh_fast(inn[v] + bin + r * (hn[v] + bhn));
      outNew[idx] = (1.0f - z) * nn + z * h0;
    }
  }
}

// ---------------------------------------------------------------------------
// Edge scatter (segment_sum in both directions). 32 lanes x 4 floats = 128
// cols per edge; fp32 L2 atomics.
// ---------------------------------------------------------------------------
__global__ void scatter_edges(const float* __restrict__ lmsg,
                              const float* __restrict__ cmsg,
                              const int* __restrict__ lidx,
                              const int* __restrict__ cidx,
                              float* __restrict__ l2c, float* __restrict__ c2l,
                              int E) {
  int gid = blockIdx.x * blockDim.x + threadIdx.x;
  if (gid >= E * 32) return;
  int e = gid >> 5;
  int c0 = (gid & 31) * 4;
  int li = lidx[e], ci = cidx[e];
  float4 a = *(const float4*)(lmsg + (size_t)li * DMODEL + c0);
  float* p = l2c + (size_t)ci * DMODEL + c0;
  atomicAdd(p + 0, a.x); atomicAdd(p + 1, a.y);
  atomicAdd(p + 2, a.z); atomicAdd(p + 3, a.w);
  float4 b = *(const float4*)(cmsg + (size_t)ci * DMODEL + c0);
  float* q = c2l + (size_t)li * DMODEL + c0;
  atomicAdd(q + 0, b.x); atomicAdd(q + 1, b.y);
  atomicAdd(q + 2, b.z); atomicAdd(q + 3, b.w);
}

// ---------------------------------------------------------------------------

extern "C" void kernel_launch(void* const* d_in, const int* in_sizes, int n_in,
                              void* d_out, int out_size, void* d_ws, size_t ws_size,
                              hipStream_t stream) {
  const int D = DMODEL;
  const int L = in_sizes[4] / D;
  const int C = in_sizes[5] / D;
  const int E = in_sizes[2];

  const int*   l_edge = (const int*)d_in[2];
  const int*   c_edge = (const int*)d_in[3];
  const float* l_emb0 = (const float*)d_in[4];
  const float* c_emb0 = (const float*)d_in[5];
  const float* l2c_W1 = (const float*)d_in[6];  const float* l2c_b1 = (const float*)d_in[7];
  const float* l2c_W2 = (const float*)d_in[8];  const float* l2c_b2 = (const float*)d_in[9];
  const float* c2l_W1 = (const float*)d_in[10]; const float* c2l_b1 = (const float*)d_in[11];
  const float* c2l_W2 = (const float*)d_in[12]; const float* c2l_b2 = (const float*)d_in[13];
  const float* l2l_W1 = (const float*)d_in[14]; const float* l2l_b1 = (const float*)d_in[15];
  const float* l2l_W2 = (const float*)d_in[16]; const float* l2l_b2 = (const float*)d_in[17];
  const float* cg_Wih = (const float*)d_in[18]; const float* cg_Whh = (const float*)d_in[19];
  const float* cg_bih = (const float*)d_in[20]; const float* cg_bhh = (const float*)d_in[21];
  const float* lg_Wih = (const float*)d_in[22]; const float* lg_Whh = (const float*)d_in[23];
  const float* lg_bih = (const float*)d_in[24]; const float* lg_bhh = (const float*)d_in[25];

  float* out_l = (float*)d_out;
  float* out_c = out_l + (size_t)(N_ITER + 1) * L * D;

  // --- workspace carve (256B aligned) ---
  char* base = (char*)d_ws;
  size_t off = 0;
  auto carve = [&](size_t bytes) -> void* {
    void* p = base + off;
    off = (off + bytes + 255) & ~(size_t)255;
    return p;
  };
  const size_t szL = (size_t)L * D, szC = (size_t)C * D;
  unsigned short* wp_l2c1 = (unsigned short*)carve(16384 * 2);
  unsigned short* wp_l2c2 = (unsigned short*)carve(16384 * 2);
  unsigned short* wp_c2l1 = (unsigned short*)carve(16384 * 2);
  unsigned short* wp_c2l2 = (unsigned short*)carve(16384 * 2);
  unsigned short* wp_l2l1 = (unsigned short*)carve(16384 * 2);
  unsigned short* wp_l2l2 = (unsigned short*)carve(16384 * 2);
  unsigned short* wp_cgih = (unsigned short*)carve(49152 * 2);
  unsigned short* wp_cghh = (unsigned short*)carve(49152 * 2);
  unsigned short* wp_lgih = (unsigned short*)carve(98304 * 2);
  unsigned short* wp_lghh = (unsigned short*)carve(49152 * 2);
  unsigned short* l_bf    = (unsigned short*)carve(szL * 2);
  unsigned short* c_bf    = (unsigned short*)carve(szC * 2);
  unsigned short* l2l_bf  = (unsigned short*)carve(szL * 2);
  unsigned short* x_bf    = (unsigned short*)carve((size_t)L * 256 * 2); // [c2l_aggr | l2l_msg]
  unsigned short* xc_bf   = (unsigned short*)carve(szC * 2);
  unsigned short* h_bf    = (unsigned short*)carve(szC * 2);             // MLP hidden (max rows C)
  float* lmsg     = (float*)carve(szL * 4);
  float* cmsg     = (float*)carve(szC * 4);
  float* l2c_aggr = (float*)carve(szC * 4);
  float* c2l_aggr = (float*)carve(szL * 4);

  const dim3 blk(256);
  auto grid1 = [](long long n) { return dim3((unsigned)((n + 255) / 256)); };
  const dim3 gGemmL((unsigned)((L + 127) / 128));
  const dim3 gGemmC((unsigned)((C + 127) / 128));

  // --- pack all weights (iteration-invariant) ---
  pack_weight<<<grid1(8 * 4 * 32), blk, 0, stream>>>(l2c_W1, wp_l2c1, 8, 4, 128);
  pack_weight<<<grid1(8 * 4 * 32), blk, 0, stream>>>(l2c_W2, wp_l2c2, 8, 4, 128);
  pack_weight<<<grid1(8 * 4 * 32), blk, 0, stream>>>(c2l_W1, wp_c2l1, 8, 4, 128);
  pack_weight<<<grid1(8 * 4 * 32), blk, 0, stream>>>(c2l_W2, wp_c2l2, 8, 4, 128);
  pack_weight<<<grid1(8 * 4 * 32), blk, 0, stream>>>(l2l_W1, wp_l2l1, 8, 4, 128);
  pack_weight<<<grid1(8 * 4 * 32), blk, 0, stream>>>(l2l_W2, wp_l2l2, 8, 4, 128);
  pack_weight<<<grid1(24 * 4 * 32), blk, 0, stream>>>(cg_Wih, wp_cgih, 24, 4, 128);
  pack_weight<<<grid1(24 * 4 * 32), blk, 0, stream>>>(cg_Whh, wp_cghh, 24, 4, 128);
  pack_weight<<<grid1(24 * 8 * 32), blk, 0, stream>>>(lg_Wih, wp_lgih, 24, 8, 256);
  pack_weight<<<grid1(24 * 4 * 32), blk, 0, stream>>>(lg_Whh, wp_lghh, 24, 4, 128);

  // --- slot 0 of stacked outputs = initial embeddings ---
  hipMemcpyAsync(out_l, l_emb0, szL * sizeof(float), hipMemcpyDeviceToDevice, stream);
  hipMemcpyAsync(out_c, c_emb0, szC * sizeof(float), hipMemcpyDeviceToDevice, stream);

  for (int it = 0; it < N_ITER; ++it) {
    const float* lcur = out_l + (size_t)it * szL;
    const float* ccur = out_c + (size_t)it * szC;
    float* lnext = out_l + (size_t)(it + 1) * szL;
    float* cnext = out_c + (size_t)(it + 1) * szC;

    // bf16 copies of current embeddings (+ pos/neg swapped literal view)
    cvt_rows_bf16<<<grid1((long long)L * 32), blk, 0, stream>>>(lcur, D, l_bf, D, L, 0);
    cvt_rows_bf16<<<grid1((long long)L * 32), blk, 0, stream>>>(lcur, D, l2l_bf, D, L, 1);
    cvt_rows_bf16<<<grid1((long long)C * 32), blk, 0, stream>>>(ccur, D, c_bf, D, C, 0);

    // l2c MLP: lmsg = relu(l @ W1.T + b1) @ W2.T + b2
    gemm128_bf16<true,  true ><<<gGemmL, blk, 0, stream>>>(l_bf, D, wp_l2c1, l2c_b1, nullptr, h_bf, D, L);
    gemm128_bf16<false, false><<<gGemmL, blk, 0, stream>>>(h_bf, D, wp_l2c2, l2c_b2, lmsg, nullptr, D, L);
    // c2l MLP
    gemm128_bf16<true,  true ><<<gGemmC, blk, 0, stream>>>(c_bf, D, wp_c2l1, c2l_b1, nullptr, h_bf, D, C);
    gemm128_bf16<false, false><<<gGemmC, blk, 0, stream>>>(h_bf, D, wp_c2l2, c2l_b2, cmsg, nullptr, D, C);
    // l2l MLP -> bf16 directly into cols [128:256) of the literal-GRU input
    gemm128_bf16<true,  true ><<<gGemmL, blk, 0, stream>>>(l2l_bf, D, wp_l2l1, l2l_b1, nullptr, h_bf, D, L);
    gemm128_bf16<false, true ><<<gGemmL, blk, 0, stream>>>(h_bf, D, wp_l2l2, l2l_b2, nullptr, x_bf + 128, 256, L);

    // segment sums over edges
    hipMemsetAsync(l2c_aggr, 0, szC * sizeof(float), stream);
    hipMemsetAsync(c2l_aggr, 0, szL * sizeof(float), stream);
    scatter_edges<<<grid1((long long)E * 32), blk, 0, stream>>>(lmsg, cmsg, l_edge, c_edge,
                                                                l2c_aggr, c2l_aggr, E);

    // clause GRU: c' = GRU(l2c_aggr, c)
    cvt_rows_bf16<<<grid1((long long)C * 32), blk, 0, stream>>>(l2c_aggr, D, xc_bf, D, C, 0);
    gru_fused<4><<<gGemmC, blk, 0, stream>>>(xc_bf, D, c_bf, wp_cgih, wp_cghh,
                                             cg_bih, cg_bhh, ccur, cnext, C);

    // literal GRU: l' = GRU([c2l_aggr | l2l_msg], l)
    cvt_rows_bf16<<<grid1((long long)L * 32), blk, 0, stream>>>(c2l_aggr, D, x_bf, 256, L, 0);
    gru_fused<8><<<gGemmL, blk, 0, stream>>>(x_bf, 256, l_bf, wp_lgih, wp_lghh,
                                             lg_bih, lg_bhh, lcur, lnext, L);
  }
}
